// GRU__618475291241
// MI455X (gfx1250) — matrix operand order repaired and checked by reading, not compiled, not run on hardware
//
#include <hip/hip_runtime.h>
#include <hip/hip_bf16.h>

#define BATCH 128
#define SEQ   1024
#define EMB   256
#define HID   512
#define KTOT  (EMB + HID)   // 768
#define VOCAB 100

// LDS slab geometry: 16 rows x 768 bf16, row padded to 776 elems (1552 B).
// 1552 B = 388 dwords; 388 mod 64 = 4 -> the 16 lanes of a B-fragment read
// land on 16 distinct 4-dword bank groups: conflict-free ds_load_b128.
#define LDS_ROW    776
#define SLAB_ELEMS (16 * LDS_ROW)          // 12416 elems = 24832 B
#define N_SLABS    12                      // 8 phase-A + 4 phase-B
// total LDS: 12 * 24832 = 297,984 B  (<= 320 KB per WGP)

typedef __attribute__((ext_vector_type(16))) __bf16 v16bf;
typedef __attribute__((ext_vector_type(8)))  float  v8f;

union V16 { v16bf v; uint4 q[2]; };

__device__ __forceinline__ unsigned short f2bf(float f) {
  unsigned int u = __float_as_uint(f);
  u += 0x7FFFu + ((u >> 16) & 1u);          // round-to-nearest-even
  return (unsigned short)(u >> 16);
}

// ---- workspace layout (bytes, all 256B aligned) ----
#define OFF_EMBBF 0u                          // 100*256 bf16      = 51200
#define OFF_WZ    51200u                      // 512*768 bf16      = 786432
#define OFF_WR    (OFF_WZ + 786432u)
#define OFF_WC    (OFF_WR + 786432u)
#define OFF_H     (OFF_WC + 786432u)          // 128*512 f32       = 262144
#define OFF_Z     (OFF_H  + 262144u)          // 128*512 f32       = 262144
#define OFF_HBF   (OFF_Z  + 262144u)          // 128*512 bf16      = 131072
#define OFF_RHBF  (OFF_HBF + 131072u)         // 128*512 bf16      = 131072
#define OFF_CTR   (OFF_RHBF + 131072u)        // barrier counter

// ------------------------------------------------------------------
// Prep: fp32 -> bf16 conversion of emb + 3 weight matrices, zero state
// ------------------------------------------------------------------
__global__ void gru_prep(const float* __restrict__ emb,
                         const float* __restrict__ Wz,
                         const float* __restrict__ Wr,
                         const float* __restrict__ Wc,
                         unsigned char* __restrict__ ws) {
  unsigned short* embbf = (unsigned short*)(ws + OFF_EMBBF);
  unsigned short* wz    = (unsigned short*)(ws + OFF_WZ);
  unsigned short* wr    = (unsigned short*)(ws + OFF_WR);
  unsigned short* wc    = (unsigned short*)(ws + OFF_WC);
  float*          h     = (float*)(ws + OFF_H);
  unsigned short* hbf   = (unsigned short*)(ws + OFF_HBF);
  unsigned short* rhbf  = (unsigned short*)(ws + OFF_RHBF);

  int g  = blockIdx.x * blockDim.x + threadIdx.x;
  int gs = gridDim.x * blockDim.x;

  for (int i = g; i < VOCAB * EMB; i += gs) embbf[i] = f2bf(emb[i]);
  for (int i = g; i < HID * KTOT; i += gs) {
    wz[i] = f2bf(Wz[i]);
    wr[i] = f2bf(Wr[i]);
    wc[i] = f2bf(Wc[i]);
  }
  for (int i = g; i < BATCH * HID; i += gs) {
    h[i] = 0.0f; hbf[i] = 0; rhbf[i] = 0;
  }
  if (g == 0) *(unsigned int*)(ws + OFF_CTR) = 0u;
}

// ------------------------------------------------------------------
// One 16x16 output tile: C += A[16xK] * B[KxN], K=768, bf16 WMMA.
// A rows: K<256 gathered embeddings (global), K>=256 from hrow (global).
// B fragment comes from the LDS-resident weight slab (loop-invariant).
// ------------------------------------------------------------------
__device__ __forceinline__ v8f tile_gemm(const unsigned short* __restrict__ embbf,
                                         const int* __restrict__ Xrow, int t,
                                         const unsigned short* __restrict__ hrow,
                                         const unsigned short* ldsB,
                                         int ksel) {
  int xrow = Xrow[t];                                     // one index per lane-row
  const unsigned short* ax = embbf + (size_t)xrow * EMB;  // x part of "combined"

  v8f c = {};
#pragma unroll
  for (int kc = 0; kc < KTOT; kc += 32) {
    const unsigned short* as = (kc < EMB) ? (ax + kc) : (hrow + (kc - EMB));
    V16 a, b;
    // A 16x32 bf16: lane holds K runs [kc+8*ksel..+7] and [kc+16+8*ksel..+7]
    a.q[0] = *(const uint4*)(as + 8 * ksel);
    a.q[1] = *(const uint4*)(as + 8 * ksel + 16);
    // B 32x16 bf16: lane holds K runs [kc+8*ksel..+7] and [kc+16+8*ksel..+7] (LDS, padded row)
    const unsigned short* bs = ldsB + kc + 8 * ksel;
    b.q[0] = *(const uint4*)(bs);
    b.q[1] = *(const uint4*)(bs + 16);
    c = __builtin_amdgcn_wmma_f32_16x16x32_bf16(false, a.v, false, b.v,
                                                (short)0, c, false, false);
  }
  return c;
}

// device-scope monotonic grid barrier
__device__ __forceinline__ void grid_barrier(unsigned int* ctr, unsigned int target) {
  __threadfence();
  __syncthreads();
  if (threadIdx.x == 0) {
    __hip_atomic_fetch_add(ctr, 1u, __ATOMIC_ACQ_REL, __HIP_MEMORY_SCOPE_AGENT);
    while (__hip_atomic_load(ctr, __ATOMIC_ACQUIRE, __HIP_MEMORY_SCOPE_AGENT) < target)
      __builtin_amdgcn_s_sleep(2);
  }
  __syncthreads();
  __threadfence();
}

// ------------------------------------------------------------------
// Persistent scan kernel: 64 WGs x 256 thr = 512 waves, loops t=0..1023.
// Per WG: 8 phase-A tiles (one per wave) + 4 phase-B tiles (waves 0-3),
// with all 12 weight slabs staged once into LDS.
// ------------------------------------------------------------------
__launch_bounds__(256, 1)
__global__ void gru_scan(const int* __restrict__ X,
                         const float* __restrict__ bz,
                         const float* __restrict__ br,
                         const float* __restrict__ bc,
                         unsigned char* __restrict__ ws,
                         float* __restrict__ out) {
  __shared__ __align__(16) unsigned short ldsW[N_SLABS * SLAB_ELEMS];

  const unsigned short* embbf = (const unsigned short*)(ws + OFF_EMBBF);
  const unsigned short* wzbf  = (const unsigned short*)(ws + OFF_WZ);
  const unsigned short* wrbf  = (const unsigned short*)(ws + OFF_WR);
  const unsigned short* wcbf  = (const unsigned short*)(ws + OFF_WC);
  float*          h    = (float*)(ws + OFF_H);
  float*          zbuf = (float*)(ws + OFF_Z);
  unsigned short* hbf  = (unsigned short*)(ws + OFF_HBF);
  unsigned short* rhbf = (unsigned short*)(ws + OFF_RHBF);
  unsigned int*   ctr  = (unsigned int*)(ws + OFF_CTR);

  // ---- stage the 12 loop-invariant weight slabs into LDS ----
  for (int idx = threadIdx.x; idx < N_SLABS * 16 * (KTOT / 8); idx += blockDim.x) {
    const int q    = idx % (KTOT / 8);      // 16-byte chunk within row
    const int j    = idx / (KTOT / 8);      // global row 0..191
    const int slab = j >> 4;
    const int r    = j & 15;
    const unsigned short* src;
    if (slab < 8) {                         // phase-A slab: tileA = blk*8 + slab
      const int tA  = blockIdx.x * 8 + slab;
      const int n0  = ((tA & 255) & 31) * 16;
      src = ((tA >> 8) ? wrbf : wzbf) + (size_t)(n0 + r) * KTOT + q * 8;
    } else {                                // phase-B slab: tileB = blk*4 + (slab-8)
      const int tB  = blockIdx.x * 4 + (slab - 8);
      const int n0  = (tB & 31) * 16;
      src = wcbf + (size_t)(n0 + r) * KTOT + q * 8;
    }
    *(uint4*)&ldsW[slab * SLAB_ELEMS + r * LDS_ROW + q * 8] = *(const uint4*)src;
  }
  __syncthreads();

  const int lane = threadIdx.x & 31;
  const int lw   = threadIdx.x >> 5;        // wave in WG, 0..7
  const int m    = lane & 15;
  const int ksel = (lane >> 4) & 1;

  // ---- per-wave loop-invariant state ----
  // Phase A: tileA = blk*8 + lw  (512 tiles: gate 0 = z, gate 1 = r)
  const int tileA = blockIdx.x * 8 + lw;
  const int gA    = tileA >> 8;
  const int b0A   = (((tileA & 255) >> 5)) * 16;
  const int n0A   = ((tileA & 255) & 31) * 16;
  const unsigned short* ldsA = &ldsW[lw * SLAB_ELEMS + m * LDS_ROW];
  const float biasA = (gA ? br : bz)[n0A + m];
  const int*  XrowA = X + (b0A + m) * SEQ;
  // Phase B: tileB = blk*4 + lw  (waves 0..3; 256 candidate tiles)
  const int tileB = blockIdx.x * 4 + (lw & 3);
  const int b0B   = (tileB >> 5) * 16;
  const int n0B   = (tileB & 31) * 16;
  const unsigned short* ldsB = &ldsW[(8 + (lw & 3)) * SLAB_ELEMS + m * LDS_ROW];
  const float biasB = bc[n0B + m];
  const int*  XrowB = X + (b0B + m) * SEQ;

  const unsigned int nb = gridDim.x;
  unsigned int bar_target = 0;

  for (int t = 0; t < SEQ; ++t) {
    // ---------- Phase A: z and r gates ----------
    {
      v8f c = tile_gemm(embbf, XrowA, t, hbf + (size_t)(b0A + m) * HID, ldsA, ksel);
      const int n = n0A + m;
#pragma unroll
      for (int v = 0; v < 8; ++v) {
        const int row = b0A + v + 8 * ksel;
        const float s = 1.0f / (1.0f + __expf(-(c[v] + biasA)));   // sigmoid
        if (gA == 0) zbuf[row * HID + n] = s;
        else         rhbf[row * HID + n] = f2bf(s * h[row * HID + n]);
      }
    }
    bar_target += nb; grid_barrier(ctr, bar_target);

    // ---------- Phase B: candidate gate + h update (waves 0..3) ----------
    if (lw < 4) {
      v8f c = tile_gemm(embbf, XrowB, t, rhbf + (size_t)(b0B + m) * HID, ldsB, ksel);
      const int n = n0B + m;
#pragma unroll
      for (int v = 0; v < 8; ++v) {
        const int row   = b0B + v + 8 * ksel;
        const float ht  = tanhf(c[v] + biasB);
        const float z   = zbuf[row * HID + n];
        const float hp  = h[row * HID + n];
        const float hn  = (1.0f - z) * hp + z * ht;
        h[row * HID + n]   = hn;
        hbf[row * HID + n] = f2bf(hn);
        out[((size_t)row * SEQ + t) * HID + n] = hn;
      }
    }
    bar_target += nb; grid_barrier(ctr, bar_target);
  }
}

extern "C" void kernel_launch(void* const* d_in, const int* in_sizes, int n_in,
                              void* d_out, int out_size, void* d_ws, size_t ws_size,
                              hipStream_t stream) {
  (void)in_sizes; (void)n_in; (void)out_size; (void)ws_size;
  const int*   X   = (const int*)d_in[0];
  const float* emb = (const float*)d_in[1];
  const float* Wz  = (const float*)d_in[2];
  const float* bz  = (const float*)d_in[3];
  const float* Wr  = (const float*)d_in[4];
  const float* br  = (const float*)d_in[5];
  const float* Wc  = (const float*)d_in[6];
  const float* bc  = (const float*)d_in[7];
  unsigned char* ws = (unsigned char*)d_ws;
  float* out = (float*)d_out;

  hipLaunchKernelGGL(gru_prep, dim3(512), dim3(256), 0, stream, emb, Wz, Wr, Wc, ws);
  hipLaunchKernelGGL(gru_scan, dim3(64), dim3(256), 0, stream, X, bz, br, bc, ws, out);
}
